// PPOInteractiveModel_52716428591857
// MI455X (gfx1250) — compile-verified, matrix operand order checked by reading, not run on hardware
//
#include <hip/hip_runtime.h>
#include <hip/hip_bf16.h>
#include <float.h>

typedef __attribute__((ext_vector_type(16))) __bf16 v16bf;
typedef __attribute__((ext_vector_type(8)))  __bf16 v8bf;
typedef __attribute__((ext_vector_type(8)))  float  v8f;

#define BB 16
#define VV 50000
#define HH 512
#define G3 1536
#define T_STEPS 32
#define TOPK 10

// ---- bf16 A/B fragment load per CDNA5 16-bit WMMA layout -------------------
// lanes 0-15  (lh=0): K = k0+0..7   (v0-3) and k0+16..23 (v4-7)
// lanes 16-31 (lh=1): K = k0+8..15  (v0-3) and k0+24..31 (v4-7)
__device__ __forceinline__ v16bf load_frag(const __bf16* __restrict__ row, int k0, int lh) {
  const v8bf lo = *reinterpret_cast<const v8bf*>(row + k0 + lh * 8);
  const v8bf hi = *reinterpret_cast<const v8bf*>(row + k0 + 16 + lh * 8);
  v16bf r;
#pragma unroll
  for (int i = 0; i < 8; ++i) { r[i] = lo[i]; r[i + 8] = hi[i]; }
  return r;
}

// ---- GEMM: C[16,N] = A[16,Kd](bf16) * Bw[N,Kd]^T(bf16) + bias[N] -----------
__global__ __launch_bounds__(256) void wmma_gemm_bf16_kernel(
    const __bf16* __restrict__ A, const __bf16* __restrict__ Bw,
    const float* __restrict__ bias, float* __restrict__ C, int N, int Kd) {
  const int wave = threadIdx.x >> 5;
  const int lane = threadIdx.x & 31;
  const int tile_n = (blockIdx.x * 8 + wave) * 16;
  const int lh = lane >> 4;
  const int ln = lane & 15;
  const int n = tile_n + ln;
  const int nc = (n < N) ? n : 0;                 // clamp loads, guard stores
  const __bf16* arow = A + (size_t)ln * Kd;       // m = ln (B=16 fits one tile)
  const __bf16* brow = Bw + (size_t)nc * Kd;
  const float bv = (n < N) ? bias[n] : 0.0f;
  v8f c;
#pragma unroll
  for (int j = 0; j < 8; ++j) c[j] = bv;          // bias folded into C init
  for (int k0 = 0; k0 < Kd; k0 += 32) {
    v16bf a = load_frag(arow, k0, lh);
    v16bf b = load_frag(brow, k0, lh);
    c = __builtin_amdgcn_wmma_f32_16x16x32_bf16(false, a, false, b,
                                                (short)0, c, false, false);
  }
  if (n < N) {
#pragma unroll
    for (int j = 0; j < 8; ++j)                   // C layout: m = j + 8*lh, n = tile_n+ln
      C[(size_t)(j + lh * 8) * N + n] = c[j];
  }
}

// ---- GRU gates (hidden == 0 -> gh = b_hh), optional value head -------------
__global__ __launch_bounds__(512) void gru_gates_kernel(
    const float* __restrict__ gi, const float* __restrict__ bhh,
    __bf16* __restrict__ xout, const float* __restrict__ value_w,
    const float* __restrict__ value_b, float* __restrict__ out_values, int t) {
  __shared__ float sred[512];
  const int b = blockIdx.x;
  const int h = threadIdx.x;
  const float gr = gi[b * G3 + h];
  const float gz = gi[b * G3 + 512 + h];
  const float gn = gi[b * G3 + 1024 + h];
  const float r = 1.0f / (1.0f + expf(-(gr + bhh[h])));
  const float z = 1.0f / (1.0f + expf(-(gz + bhh[512 + h])));
  const float nn = tanhf(gn + r * bhh[1024 + h]);
  const float out = (1.0f - z) * nn;              // + z*h, h==0
  xout[b * HH + h] = (__bf16)out;
  if (value_w != nullptr) {
    sred[h] = out * value_w[h];
    __syncthreads();
    for (int s = 256; s > 0; s >>= 1) {
      if (h < s) sred[h] += sred[h + s];
      __syncthreads();
    }
    if (h == 0) out_values[b * T_STEPS + t] = sred[0] + value_b[0];
  }
}

// ---- fused softmax + topK + hit/select + history + next input --------------
__global__ __launch_bounds__(1024) void softmax_select_kernel(
    const float* __restrict__ logits, float* __restrict__ history,
    const float* __restrict__ interesting, const float* __restrict__ embedding,
    float* __restrict__ out_logits, float* __restrict__ out_sample,
    float* __restrict__ out_hits, __bf16* __restrict__ x_in, int t) {
  const int STPB = 1024;
  __shared__ float s_red[1024];
  __shared__ int   s_idx[1024];
  __shared__ float s_pol[1024];
  __shared__ float s_topm[TOPK], s_topp[TOPK];
  __shared__ int   s_topi[TOPK];
  __shared__ int   s_sel;
  __shared__ float s_hff;

  const int b = blockIdx.x;
  const int tid = threadIdx.x;
  const float* lrow = logits + (size_t)b * VV;
  const float* hrow = history + (size_t)b * VV;

  // pass 1: row max
  float m = -FLT_MAX;
  for (int v = tid; v < VV; v += STPB) m = fmaxf(m, lrow[v]);
  s_red[tid] = m;
  __syncthreads();
  for (int s = STPB / 2; s > 0; s >>= 1) {
    if (tid < s) s_red[tid] = fmaxf(s_red[tid], s_red[tid + s]);
    __syncthreads();
  }
  const float rmax = s_red[0];
  __syncthreads();

  // pass 2: sum of exp
  float sum = 0.f;
  for (int v = tid; v < VV; v += STPB) sum += expf(lrow[v] - rmax);
  s_red[tid] = sum;
  __syncthreads();
  for (int s = STPB / 2; s > 0; s >>= 1) {
    if (tid < s) s_red[tid] += s_red[tid + s];
    __syncthreads();
  }
  const float inv = 1.0f / s_red[0];
  __syncthreads();

  // pass 3: write policy, per-thread top-10 of masked = policy*history
  float bm[TOPK], bp[TOPK];
  int bi[TOPK];
#pragma unroll
  for (int j = 0; j < TOPK; ++j) { bm[j] = -FLT_MAX; bp[j] = 0.f; bi[j] = 0x7fffffff; }
  float* orow = out_logits + ((size_t)b * T_STEPS + t) * VV;
  for (int v = tid; v < VV; v += STPB) {
    const float p = expf(lrow[v] - rmax) * inv;
    orow[v] = p;
    const float mk = p * hrow[v];
    if (mk > bm[TOPK - 1] || (mk == bm[TOPK - 1] && v < bi[TOPK - 1])) {
      bm[TOPK - 1] = mk; bp[TOPK - 1] = p; bi[TOPK - 1] = v;
#pragma unroll
      for (int j = TOPK - 1; j > 0; --j) {
        const bool sw = (bm[j] > bm[j - 1]) || (bm[j] == bm[j - 1] && bi[j] < bi[j - 1]);
        if (sw) {
          float tm = bm[j]; bm[j] = bm[j - 1]; bm[j - 1] = tm;
          float tp = bp[j]; bp[j] = bp[j - 1]; bp[j - 1] = tp;
          int   ti = bi[j]; bi[j] = bi[j - 1]; bi[j - 1] = ti;
        }
      }
    }
  }
  __syncthreads();

  // merge: TOPK rounds of block arg-max over per-thread heads
  int pos = 0;
  for (int k = 0; k < TOPK; ++k) {
    const float cm = (pos < TOPK) ? bm[pos] : -FLT_MAX;
    const int   ci = (pos < TOPK) ? bi[pos] : 0x7fffffff;
    const float cp = (pos < TOPK) ? bp[pos] : 0.f;
    s_red[tid] = cm; s_idx[tid] = ci; s_pol[tid] = cp;
    __syncthreads();
    for (int s = STPB / 2; s > 0; s >>= 1) {
      if (tid < s) {
        const bool bt = (s_red[tid + s] > s_red[tid]) ||
                        (s_red[tid + s] == s_red[tid] && s_idx[tid + s] < s_idx[tid]);
        if (bt) { s_red[tid] = s_red[tid + s]; s_idx[tid] = s_idx[tid + s]; s_pol[tid] = s_pol[tid + s]; }
      }
      __syncthreads();
    }
    if (tid == 0) { s_topm[k] = s_red[0]; s_topi[k] = s_idx[0]; s_topp[k] = s_pol[0]; }
    __syncthreads();
    if (pos < TOPK && ci == s_topi[k]) ++pos;   // winner advances its head
    __syncthreads();
  }

  // selection logic (tiny, serial on lane 0)
  if (tid == 0) {
    float hff = 0.f;
    int sel = s_topi[0];
    float bestp = -FLT_MAX; int besti = 0x7fffffff;
    for (int k = 0; k < TOPK; ++k) {
      const int idx = s_topi[k];
      if (idx < VV && interesting[(size_t)b * VV + idx] > 0.f) {
        hff = 1.f;
        const float p = s_topp[k];
        if (p > bestp || (p == bestp && idx < besti)) { bestp = p; besti = idx; }
      }
    }
    if (hff > 0.f) sel = besti;                 // else: argmax(masked) == top1
    out_sample[b * T_STEPS + t] = (float)sel;
    out_hits[b * T_STEPS + t] = hff;
    if (hff > 0.f) history[(size_t)b * VV + sel] -= 1.0f;
    s_sel = sel;
    s_hff = hff;
  }
  __syncthreads();

  // next input = (2*hff - 1) * embedding[sel]
  const float sc = 2.0f * s_hff - 1.0f;
  const float* erow = embedding + (size_t)s_sel * HH;
  for (int h = tid; h < HH; h += STPB) x_in[b * HH + h] = (__bf16)(sc * erow[h]);
}

// ---- misc helpers ----------------------------------------------------------
__global__ void f32_to_bf16_kernel(const float* __restrict__ src,
                                   __bf16* __restrict__ dst, int n) {
  for (int i = blockIdx.x * blockDim.x + threadIdx.x; i < n; i += gridDim.x * blockDim.x)
    dst[i] = (__bf16)src[i];
}

__global__ void init_kernel(const float* __restrict__ masking,
                            float* __restrict__ history,
                            const float* __restrict__ embedding,
                            __bf16* __restrict__ x_in) {
  const int total = BB * VV;
  for (int i = blockIdx.x * blockDim.x + threadIdx.x; i < total; i += gridDim.x * blockDim.x) {
    history[i] = masking[i];
    if (i < BB * HH) x_in[i] = (__bf16)embedding[i & (HH - 1)];  // emb[start_token=0]
  }
}

__global__ void copy_kernel(const float* __restrict__ src, float* __restrict__ dst, int n) {
  for (int i = blockIdx.x * blockDim.x + threadIdx.x; i < n; i += gridDim.x * blockDim.x)
    dst[i] = src[i];
}

// ---- host side -------------------------------------------------------------
extern "C" void kernel_launch(void* const* d_in, const int* in_sizes, int n_in,
                              void* d_out, int out_size, void* d_ws, size_t ws_size,
                              hipStream_t stream) {
  (void)in_sizes; (void)n_in; (void)out_size; (void)ws_size;
  const float* interesting = (const float*)d_in[0];
  const float* masking     = (const float*)d_in[1];
  const float* embedding   = (const float*)d_in[2];
  const float* gw_ih       = (const float*)d_in[3];
  /* d_in[4] gru_w_hh: unused (hidden state is always zero in the reference) */
  const float* gb_ih       = (const float*)d_in[5];
  const float* gb_hh       = (const float*)d_in[6];
  const float* fc_w        = (const float*)d_in[7];
  const float* fc_b        = (const float*)d_in[8];
  const float* value_w     = (const float*)d_in[9];
  const float* value_b     = (const float*)d_in[10];
  /* d_in[11] sequence_length == 32 (compile-time constant T_STEPS) */

  float* out = (float*)d_out;
  float* out_logits = out;                                     // [B,T,V]
  float* out_sample = out_logits + (size_t)BB * T_STEPS * VV;  // [B,T]
  float* out_values = out_sample + BB * T_STEPS;               // [B,T,1]
  float* out_fhist  = out_values + BB * T_STEPS;               // [B,V]
  float* out_hits   = out_fhist + (size_t)BB * VV;             // [B,T,1]

  char* ws = (char*)d_ws;
  auto take = [&](size_t bytes) { char* p = ws; ws += (bytes + 255) & ~(size_t)255; return p; };
  __bf16* fcw16   = (__bf16*)take((size_t)VV * HH * 2);    // 51.2 MB, L2-resident
  __bf16* wih16   = (__bf16*)take((size_t)2 * G3 * HH * 2);
  float*  logits  = (float*) take((size_t)BB * VV * 4);
  float*  history = (float*) take((size_t)BB * VV * 4);
  float*  gi      = (float*) take((size_t)BB * G3 * 4);
  __bf16* x_in    = (__bf16*)take((size_t)BB * HH * 2);
  __bf16* x_mid   = (__bf16*)take((size_t)BB * HH * 2);
  __bf16* x_top   = (__bf16*)take((size_t)BB * HH * 2);

  f32_to_bf16_kernel<<<2048, 256, 0, stream>>>(fc_w, fcw16, VV * HH);
  f32_to_bf16_kernel<<<512, 256, 0, stream>>>(gw_ih, wih16, 2 * G3 * HH);
  init_kernel<<<1024, 256, 0, stream>>>(masking, history, embedding, x_in);

  const int fc_grid = (VV + 127) / 128;   // 8 waves * 16 cols per block = 391
  const int gi_grid = G3 / 128;           // 12

  for (int t = 0; t < T_STEPS; ++t) {
    wmma_gemm_bf16_kernel<<<gi_grid, 256, 0, stream>>>(x_in, wih16, gb_ih, gi, G3, HH);
    gru_gates_kernel<<<BB, 512, 0, stream>>>(gi, gb_hh, x_mid, nullptr, nullptr, nullptr, t);
    wmma_gemm_bf16_kernel<<<gi_grid, 256, 0, stream>>>(x_mid, wih16 + (size_t)G3 * HH,
                                                       gb_ih + G3, gi, G3, HH);
    gru_gates_kernel<<<BB, 512, 0, stream>>>(gi, gb_hh + G3, x_top, value_w, value_b,
                                             out_values, t);
    wmma_gemm_bf16_kernel<<<fc_grid, 256, 0, stream>>>(x_top, fcw16, fc_b, logits, VV, HH);
    softmax_select_kernel<<<BB, 1024, 0, stream>>>(logits, history, interesting, embedding,
                                                   out_logits, out_sample, out_hits, x_in, t);
  }
  copy_kernel<<<1024, 256, 0, stream>>>(history, out_fhist, BB * VV);
}
